// Network_40604620817206
// MI455X (gfx1250) — compile-verified, matrix operand order checked by reading, not run on hardware
//
#include <hip/hip_runtime.h>

// ---------------------------------------------------------------------------
// FNO network on MI455X (gfx1250).
//  * All heavy math on v_wmma_f32_16x16x32_bf16 (f32 accumulate).
//  * Truncated rfft/irfft along 264-axes recast as DFT-basis GEMMs.
//  * All bf16 weight tiles moved global->LDS by the Tensor Data Mover
//    (tensor_load_to_lds, TENSORcnt), with TDM-inserted LDS row padding.
//  * f32 activations staged with vectorized loads + packed bf16 LDS stores.
// ---------------------------------------------------------------------------

#define HP 264          // padded spatial size (256 + 8)
#define CW 64           // channel width
#define BASLD 320       // padded K extent of forward DFT basis rows
#define NPIX (4 * HP * HP)          // 278784 pixels
#define ACT_ELEMS (NPIX * CW)       // 17,842,176 floats per activation
#define SLICES (4 * HP)             // 1056 row/col slices per direction
#define HSTRIDE (HP * CW)           // 16896

typedef __attribute__((ext_vector_type(16))) __bf16 bf16x16;
typedef __attribute__((ext_vector_type(8)))  float  floatx8;
typedef __attribute__((ext_vector_type(4)))  unsigned int u32x4;
typedef __attribute__((ext_vector_type(8)))  int i32x8;
typedef __attribute__((ext_vector_type(4)))  int i32x4;

__device__ __forceinline__ __bf16 f2bf(float f) {
  union { float f; unsigned u; } a; a.f = f;
  unsigned r = a.u + 0x7fffu + ((a.u >> 16) & 1u);   // round-to-nearest-even
  union { unsigned short s; __bf16 b; } o;
  o.s = (unsigned short)(r >> 16);
  return o.b;
}

__device__ __forceinline__ unsigned pack2bf(float a, float b) {
  union { float f; unsigned u; } x;
  x.f = a; unsigned lo = (x.u + 0x7fffu + ((x.u >> 16) & 1u)) >> 16;
  x.f = b; unsigned hi = (x.u + 0x7fffu + ((x.u >> 16) & 1u)) >> 16;
  return (lo & 0xffffu) | (hi << 16);
}

// Column index adjust for TDM-padded LDS tiles: TDM inserts 1 dword (2 bf16)
// after every 16 dwords (32 bf16) written.
__device__ __forceinline__ int padcol(int k) { return k + ((k >> 5) << 1); }

// Load a 16x32 bf16 WMMA fragment from LDS (row-major [16][ld]; also used for
// B tiles staged column-major). Pairs (k,k+1) contiguous -> ds_load_2addr_b32.
__device__ __forceinline__ bf16x16 load_frag(const __bf16* p, int ld, int lane) {
  int L = lane & 15, g = lane >> 4;
  bf16x16 r;
#pragma unroll
  for (int v = 0; v < 8; ++v) {
    int k = ((v & 3) << 1) + (g << 3) + ((v >> 2) << 4);
    r[2 * v]     = p[L * ld + k];
    r[2 * v + 1] = p[L * ld + k + 1];
  }
  return r;
}

#define WMMA_BF16(A, B, C) \
  __builtin_amdgcn_wmma_f32_16x16x32_bf16(false, (A), false, (B), (short)0, (C), false, false)

// ---------------------------------------------------------------------------
// Tensor Data Mover: 2D tile (rows x rowlen bf16, row stride stride_elems) from
// global to LDS, with pad: +1 dword per 16 dwords -> LDS ld = rowlen + rowlen/16.
// D# layout per CDNA5 ISA ch.8 (group0 128b, group1 256b; groups 2/3 unused/0).
// This toolchain's builtin is the 6-arg form (extra int32x8 before cpol).
// ---------------------------------------------------------------------------
__device__ __forceinline__ void tdm_load_2d(const __bf16* gsrc, unsigned lds_off,
                                            int rows, int rowlen, int stride_elems) {
  unsigned long long ga = (unsigned long long)gsrc;
  u32x4 g0;
  g0[0] = 1u;                                               // count=1, user desc
  g0[1] = lds_off;                                          // lds_addr (bytes)
  g0[2] = (unsigned)ga;                                     // global_addr[31:0]
  g0[3] = (unsigned)((ga >> 32) & 0x01FFFFFFull) | 0x80000000u; // [56:32] | type=2
  i32x8 g1;
  g1[0] = (1 << 16) | (1 << 20) | (3 << 22);  // data_size=2B, pad_en, ivl=16dw, amt=1dw
  g1[1] = (rowlen & 0xFFFF) << 16;                          // tensor_dim0 lo16
  g1[2] = ((rowlen >> 16) & 0xFFFF) | ((rows & 0xFFFF) << 16); // dim0 hi | dim1 lo
  g1[3] = ((rows >> 16) & 0xFFFF) | ((rowlen & 0xFFFF) << 16); // dim1 hi | tile_dim0
  g1[4] = rows & 0xFFFF;                                    // tile_dim1 (tile_dim2=0)
  g1[5] = stride_elems;                                     // dim0_stride lo32
  g1[6] = 0;
  g1[7] = 0;
  i32x4 z4 = {0, 0, 0, 0};
  i32x8 z8 = {0, 0, 0, 0, 0, 0, 0, 0};
  __builtin_amdgcn_tensor_load_to_lds(g0, g1, z4, z4, z8, 0);
}

__device__ __forceinline__ unsigned lds_addr_of(const void* p) {
  return (unsigned)(unsigned long long)p;   // low 32 bits of flat ptr = LDS offset
}

// ---------------------------------------------------------------------------
// Weight / basis preparation (cheap, rebuilt every launch: deterministic).
// All GEMM B-operands are stored pre-transposed (col-major) so TDM rows are
// contiguous in global memory.
// ---------------------------------------------------------------------------
__global__ void k_build_basis(__bf16* bas, __bf16* basinvT) {
  const float invs = 1.0f / sqrtf(264.0f);
  // forward basis, row-major [m=64][BASLD], used as WMMA A operand
  for (int e = threadIdx.x; e < 64 * BASLD; e += blockDim.x) {
    int m = e / BASLD, w = e % BASLD;
    int k = m >> 1, im = m & 1;
    float v = 0.0f;
    if (w < 264) {
      float th = 6.28318530717958647f * (float)k * (float)w / 264.0f;
      v = (im ? -__sinf(th) : __cosf(th)) * invs;
    }
    bas[e] = f2bf(v);
  }
  // inverse basis, transposed [w=320][m=64], WMMA B operand (TDM rows = w)
  for (int e = threadIdx.x; e < BASLD * 64; e += blockDim.x) {
    int w = e >> 6, m = e & 63;
    int k = m >> 1, im = m & 1;
    float v = 0.0f;
    if (w < 264) {
      float th = 6.28318530717958647f * (float)k * (float)w / 264.0f;
      v = (im ? -__sinf(th) : __cosf(th)) * invs * (k == 0 ? 1.0f : 2.0f);
    }
    basinvT[e] = f2bf(v);
  }
}

// Complex per-mode mix weights (64,64,32,2) -> 32 col-major 128x128 matrices:
// mixT[k][n][p] = Wk[p][n], Wk = [[Re, Im], [-Im, Re]] on packed [Re_i, Im_i].
__global__ void k_pack_mix(const float* __restrict__ wsrc, __bf16* __restrict__ mixT) {
  int k = blockIdx.x;
  for (int e = threadIdx.x; e < 128 * 128; e += blockDim.x) {
    int n = e >> 7, p = e & 127;
    int i = p & 63, o = n & 63;
    const float* b = wsrc + (((i * 64 + o) * 32 + k) << 1);
    float re = b[0], imv = b[1];
    float v = (p < 64) ? ((n < 64) ? re : imv) : ((n < 64) ? -imv : re);
    mixT[k * 16384 + e] = f2bf(v);
  }
}

// Transpose-pack: dst[n*K + k] = src[k*N + n] (col-major bf16 for TDM).
__global__ void k_pack_wT(const float* __restrict__ s, __bf16* __restrict__ d,
                          int K, int N) {
  int i = blockIdx.x * blockDim.x + threadIdx.x;
  if (i >= K * N) return;
  int n = i / K, k = i % K;
  d[n * K + k] = f2bf(s[k * N + n]);
}

// ---------------------------------------------------------------------------
// Encode: BHWC linear (cin+2 -> 64) with coordinate grid, zero pad to 264x264
// ---------------------------------------------------------------------------
__global__ void k_encode(const float* __restrict__ inp, int cin, int bstride,
                         const float* __restrict__ ew, const float* __restrict__ eb,
                         float* __restrict__ out) {
  int p = blockIdx.x * blockDim.x + threadIdx.x;
  if (p >= NPIX) return;
  int b = p / (HP * HP), r = p % (HP * HP);
  int h = r / HP, w = r % HP;
  float* o = out + (long)p * CW;
  if (h < 256 && w < 256) {
    float f[8];
    const float* ib = inp + (long)b * bstride + (h * 256 + w);
#pragma unroll 5
    for (int c = 0; c < cin; ++c) f[c] = ib[c * 65536];
    f[cin]     = (float)h * (1.0f / 255.0f);
    f[cin + 1] = (float)w * (1.0f / 255.0f);
    for (int oc = 0; oc < CW; ++oc) {
      float acc = eb[oc];
      for (int c = 0; c < cin + 2; ++c) acc += f[c] * ew[c * CW + oc];
      o[oc] = acc;
    }
  } else {
    for (int oc = 0; oc < CW; ++oc) o[oc] = 0.0f;
  }
}

// ---------------------------------------------------------------------------
// Forward DFT: per slice s, S1[s](64 modes x 64 chan) = Bas(64x288) @ Act(288x64).
// Basis tile (64x288, ld 306 padded) loaded once by TDM; activation staged
// per-K-step with float4 loads + bf16 conversion.
// ---------------------------------------------------------------------------
__global__ void __launch_bounds__(512)
k_dft_fwd(const float* __restrict__ act, const __bf16* __restrict__ bas,
          float* __restrict__ s1, int dirH) {
  __shared__ __align__(16) __bf16 AsW[64 * 306];   // TDM-padded basis
  __shared__ __align__(16) __bf16 Bs[64][34];
  int s = blockIdx.x;
  long base; int strideK;
  if (dirH) { int b = s / HP, w = s % HP; base = (long)b * (HP * HP * CW) + (long)w * CW; strideK = HSTRIDE; }
  else      { base = (long)s * HSTRIDE; strideK = CW; }
  int tid = threadIdx.x, lane = tid & 31, wv = tid >> 5;
  int wm = wv >> 2, wn = wv & 3;
  if (tid < 32) {
    tdm_load_2d(bas, lds_addr_of(AsW), 64, 288, BASLD);
    __builtin_amdgcn_s_wait_tensorcnt(0);
  }
  floatx8 acc = {};
  int n4 = (tid & 15) << 2, kk = tid >> 4;
  for (int kt = 0; kt < 9; ++kt) {
    int kg = kt * 32 + kk;
    float4 v = {0.f, 0.f, 0.f, 0.f};
    if (kg < 264) v = *(const float4*)&act[base + (long)kg * strideK + n4];
    __syncthreads();            // also publishes TDM tile on first iteration
    Bs[n4 + 0][kk] = f2bf(v.x);
    Bs[n4 + 1][kk] = f2bf(v.y);
    Bs[n4 + 2][kk] = f2bf(v.z);
    Bs[n4 + 3][kk] = f2bf(v.w);
    __syncthreads();
    bf16x16 fa = load_frag(AsW + (wm * 16) * 306 + padcol(kt * 32), 306, lane);
    bf16x16 fb = load_frag(&Bs[wn * 16][0], 34, lane);
    acc = WMMA_BF16(fa, fb, acc);
  }
  int g = lane >> 4, n = wn * 16 + (lane & 15);
#pragma unroll
  for (int j = 0; j < 8; ++j) {
    int m = wm * 16 + j + 8 * g;
    s1[(long)s * 4096 + m * CW + n] = acc[j];
  }
}

// ---------------------------------------------------------------------------
// Per-mode complex channel mix: rows = 1056 slices, 128->128 GEMM per mode k.
// Mix matrix (128x128, ld 136 padded) loaded whole by TDM.
// ---------------------------------------------------------------------------
__global__ void __launch_bounds__(512)
k_mix(const float* __restrict__ s1, const __bf16* __restrict__ mixT,
      float* __restrict__ s2) {
  __shared__ __align__(16) __bf16 As[32][34];
  __shared__ __align__(16) __bf16 BsW[128 * 136];
  int m0 = blockIdx.x * 32;
  int k = blockIdx.y;
  int tid = threadIdx.x, lane = tid & 31, wv = tid >> 5;
  int wm = wv >> 3, wn = wv & 7;
  if (tid < 32) {
    tdm_load_2d(mixT + k * 16384, lds_addr_of(BsW), 128, 128, 128);
    __builtin_amdgcn_s_wait_tensorcnt(0);
  }
  floatx8 acc = {};
  int r = tid >> 4, kk2 = (tid & 15) << 1;
  for (int kt = 0; kt < 4; ++kt) {
    float2 v = *(const float2*)&s1[(long)(m0 + r) * 4096 + k * 128 + kt * 32 + kk2];
    __syncthreads();
    *(unsigned*)&As[r][kk2] = pack2bf(v.x, v.y);
    __syncthreads();
    bf16x16 fa = load_frag(&As[wm * 16][0], 34, lane);
    bf16x16 fb = load_frag(BsW + (wn * 16) * 136 + padcol(kt * 32), 136, lane);
    acc = WMMA_BF16(fa, fb, acc);
  }
  int g = lane >> 4, n = wn * 16 + (lane & 15);
#pragma unroll
  for (int j = 0; j < 8; ++j) {
    int m = m0 + wm * 16 + j + 8 * g;
    s2[(long)m * 4096 + k * 128 + n] = acc[j];
  }
}

// ---------------------------------------------------------------------------
// Inverse DFT, transposed: Out^T(64 chan x 64 pos-tile) = S2^T(64x64) @ BasInvT.
// BasInvT tile (64 pos rows x 64 modes, ld 68) loaded whole by TDM.
// ---------------------------------------------------------------------------
__global__ void __launch_bounds__(512)
k_dft_inv(const float* __restrict__ s2, const __bf16* __restrict__ basinvT,
          float* __restrict__ outp, int dirH, int accum) {
  __shared__ __align__(16) __bf16 As[64][34];
  __shared__ __align__(16) __bf16 BsW[64 * 68];
  int s = blockIdx.x;
  int w0 = blockIdx.y * 64;
  long base; int strideK;
  if (dirH) { int b = s / HP, w = s % HP; base = (long)b * (HP * HP * CW) + (long)w * CW; strideK = HSTRIDE; }
  else      { base = (long)s * HSTRIDE; strideK = CW; }
  int tid = threadIdx.x, lane = tid & 31, wv = tid >> 5;
  int wm = wv >> 2, wn = wv & 3;
  if (tid < 32) {
    tdm_load_2d(basinvT + (long)w0 * 64, lds_addr_of(BsW), 64, 64, 64);
    __builtin_amdgcn_s_wait_tensorcnt(0);
  }
  floatx8 acc = {};
  int c4 = (tid & 15) << 2, kk = tid >> 4;
  for (int kt = 0; kt < 2; ++kt) {
    float4 v = *(const float4*)&s2[(long)s * 4096 + (kt * 32 + kk) * CW + c4];
    __syncthreads();
    As[c4 + 0][kk] = f2bf(v.x);
    As[c4 + 1][kk] = f2bf(v.y);
    As[c4 + 2][kk] = f2bf(v.z);
    As[c4 + 3][kk] = f2bf(v.w);
    __syncthreads();
    bf16x16 fa = load_frag(&As[wm * 16][0], 34, lane);
    bf16x16 fb = load_frag(BsW + (wn * 16) * 68 + padcol(kt * 32), 68, lane);
    acc = WMMA_BF16(fa, fb, acc);
  }
  int g = lane >> 4;
  int wloc = w0 + wn * 16 + (lane & 15);
  if (wloc < 264) {
#pragma unroll
    for (int j = 0; j < 8; ++j) {
      int c = wm * 16 + j + 8 * g;
      long a = base + (long)wloc * strideK + c;
      float v = acc[j];
      if (accum) v += outp[a];
      outp[a] = v;
    }
  }
}

// ---------------------------------------------------------------------------
// Fused MLP: T += m2( relu(m1(Z) + b1) ) + b2; hidden 64x128 lives in LDS.
// Both weight matrices resident in LDS via TDM (col-major, TDM-padded).
// ---------------------------------------------------------------------------
__global__ void __launch_bounds__(512)
k_mlp(const float* __restrict__ z,
      const __bf16* __restrict__ w1T, const float* __restrict__ b1,
      const __bf16* __restrict__ w2T, const float* __restrict__ b2,
      float* __restrict__ t) {
  __shared__ __align__(16) __bf16 As[64 * 66];     // z tile, VALU staged
  __shared__ __align__(16) __bf16 Bs1[128 * 68];   // w1T via TDM (128 rows x 64)
  __shared__ __align__(16) __bf16 Hs[64 * 130];    // hidden bf16
  __shared__ __align__(16) __bf16 Bs2[64 * 136];   // w2T via TDM (64 rows x 128)
  long p0 = (long)blockIdx.x * 64;
  int tid = threadIdx.x, lane = tid & 31, wv = tid >> 5;
  int wm = wv >> 2, wn = wv & 3;
  if (tid < 32) {
    tdm_load_2d(w1T, lds_addr_of(Bs1), 128, 64, 64);
    tdm_load_2d(w2T, lds_addr_of(Bs2), 64, 128, 128);
    __builtin_amdgcn_s_wait_tensorcnt(0);
  }
#pragma unroll
  for (int i = 0; i < 4; ++i) {
    int e = tid + i * 512;
    int r = e >> 5, c2 = (e & 31) << 1;
    float2 v = *(const float2*)&z[(p0 + r) * CW + c2];
    *(unsigned*)&As[r * 66 + c2] = pack2bf(v.x, v.y);
  }
  __syncthreads();
  floatx8 a0 = {}, a1 = {};
#pragma unroll
  for (int kt = 0; kt < 2; ++kt) {
    int ks = kt * 32;
    bf16x16 fa  = load_frag(As + (wm * 16) * 66 + ks, 66, lane);
    bf16x16 fb0 = load_frag(Bs1 + (wn * 16) * 68 + padcol(ks), 68, lane);
    bf16x16 fb1 = load_frag(Bs1 + ((wn + 4) * 16) * 68 + padcol(ks), 68, lane);
    a0 = WMMA_BF16(fa, fb0, a0);
    a1 = WMMA_BF16(fa, fb1, a1);
  }
  int g = lane >> 4;
  int n0 = wn * 16 + (lane & 15), n1 = n0 + 64;
#pragma unroll
  for (int j = 0; j < 8; ++j) {
    int m = wm * 16 + j + 8 * g;
    float h0 = a0[j] + b1[n0]; h0 = h0 > 0.0f ? h0 : 0.0f;
    float h1 = a1[j] + b1[n1]; h1 = h1 > 0.0f ? h1 : 0.0f;
    Hs[m * 130 + n0] = f2bf(h0);
    Hs[m * 130 + n1] = f2bf(h1);
  }
  __syncthreads();
  floatx8 acc = {};
#pragma unroll
  for (int kt = 0; kt < 4; ++kt) {
    int ks = kt * 32;
    bf16x16 fa = load_frag(Hs + (wm * 16) * 130 + ks, 130, lane);
    bf16x16 fb = load_frag(Bs2 + (wn * 16) * 136 + padcol(ks), 136, lane);
    acc = WMMA_BF16(fa, fb, acc);
  }
  int n = wn * 16 + (lane & 15);
#pragma unroll
  for (int j = 0; j < 8; ++j) {
    int m = wm * 16 + j + 8 * g;
    long a = (p0 + m) * CW + n;
    t[a] += acc[j] + b2[n];
  }
}

// ---------------------------------------------------------------------------
// Fuse: Z = [a+g | a-g | a*g](192) @ fuse_w(192x64) + fuse_b.
// Concat features built in LDS; fuse_wT (64 rows x 192, ld 204) via TDM.
// ---------------------------------------------------------------------------
__global__ void __launch_bounds__(512)
k_fuse(const float* __restrict__ a, const float* __restrict__ gact,
       const __bf16* __restrict__ fwT, const float* __restrict__ bias,
       float* __restrict__ outp) {
  __shared__ __align__(16) __bf16 As[64][194];
  __shared__ __align__(16) __bf16 BsW[64 * 204];
  long p0 = (long)blockIdx.x * 64;
  int tid = threadIdx.x, lane = tid & 31, wv = tid >> 5;
  int wm = wv >> 2, wn = wv & 3;
  if (tid < 32) {
    tdm_load_2d(fwT, lds_addr_of(BsW), 64, 192, 192);
    __builtin_amdgcn_s_wait_tensorcnt(0);
  }
#pragma unroll
  for (int i = 0; i < 4; ++i) {
    int e = tid + i * 512;
    int r = e >> 5, c2 = (e & 31) << 1;
    float2 av = *(const float2*)&a[(p0 + r) * CW + c2];
    float2 gv = *(const float2*)&gact[(p0 + r) * CW + c2];
    *(unsigned*)&As[r][c2]       = pack2bf(av.x + gv.x, av.y + gv.y);
    *(unsigned*)&As[r][64 + c2]  = pack2bf(av.x - gv.x, av.y - gv.y);
    *(unsigned*)&As[r][128 + c2] = pack2bf(av.x * gv.x, av.y * gv.y);
  }
  __syncthreads();
  floatx8 acc = {};
#pragma unroll
  for (int kt = 0; kt < 6; ++kt) {
    int ks = kt * 32;
    bf16x16 fa = load_frag(&As[wm * 16][ks], 194, lane);
    bf16x16 fb = load_frag(BsW + (wn * 16) * 204 + padcol(ks), 204, lane);
    acc = WMMA_BF16(fa, fb, acc);
  }
  int g = lane >> 4, n = wn * 16 + (lane & 15);
#pragma unroll
  for (int j = 0; j < 8; ++j) {
    int m = wm * 16 + j + 8 * g;
    outp[(p0 + m) * CW + n] = acc[j] + bias[n];
  }
}

// ---------------------------------------------------------------------------
// Projection: slice off pad, 64->3, transpose to (B,3,256,256). Tiny N -> VALU.
// ---------------------------------------------------------------------------
__global__ void k_project(const float* __restrict__ z, const float* __restrict__ pw,
                          const float* __restrict__ pb, float* __restrict__ outp) {
  int p = blockIdx.x * blockDim.x + threadIdx.x;
  if (p >= 4 * 256 * 256) return;
  int b = p >> 16, r = p & 65535, h = r >> 8, w = r & 255;
  const float* zr = z + ((long)(b * HP + h) * HP + w) * CW;
#pragma unroll
  for (int c = 0; c < 3; ++c) {
    float acc = pb[c];
    for (int d = 0; d < CW; ++d) acc += zr[d] * pw[c * CW + d];
    outp[(long)(b * 3 + c) * 65536 + (h << 8) + w] = acc;
  }
}

// ---------------------------------------------------------------------------
// Host orchestration
// ---------------------------------------------------------------------------
extern "C" void kernel_launch(void* const* d_in, const int* in_sizes, int n_in,
                              void* d_out, int out_size, void* d_ws, size_t ws_size,
                              hipStream_t stream) {
  const float* x        = (const float*)d_in[0];
  const float* ug       = (const float*)d_in[1];
  const float* src_w    = (const float*)d_in[2];
  const float* src_b    = (const float*)d_in[3];
  const float* geo_w    = (const float*)d_in[4];
  const float* geo_b    = (const float*)d_in[5];
  const float* front_wy = (const float*)d_in[6];
  const float* front_wx = (const float*)d_in[7];
  const float* front_m1w= (const float*)d_in[8];
  const float* front_m1b= (const float*)d_in[9];
  const float* front_m2w= (const float*)d_in[10];
  const float* front_m2b= (const float*)d_in[11];
  const float* back_wy  = (const float*)d_in[12];
  const float* back_wx  = (const float*)d_in[13];
  const float* back_m1w = (const float*)d_in[14];
  const float* back_m1b = (const float*)d_in[15];
  const float* back_m2w = (const float*)d_in[16];
  const float* back_m2b = (const float*)d_in[17];
  const float* fuse_w   = (const float*)d_in[18];
  const float* fuse_b   = (const float*)d_in[19];
  const float* proj_w   = (const float*)d_in[20];
  const float* proj_b   = (const float*)d_in[21];

  // Workspace carve (~240 MB).
  float* A  = (float*)d_ws;
  float* G  = A + ACT_ELEMS;
  float* Z  = G + ACT_ELEMS;
  float* S1 = Z + ACT_ELEMS;
  float* S2 = S1 + SLICES * 4096;
  __bf16* Bas     = (__bf16*)(S2 + SLICES * 4096);
  __bf16* BasInvT = Bas + 64 * BASLD;
  __bf16* MixY    = BasInvT + BASLD * 64;
  __bf16* MixX    = MixY + 32 * 16384;
  __bf16* M1T     = MixX + 32 * 16384;
  __bf16* M2T     = M1T + 8192;
  __bf16* FWT     = M2T + 8192;

  const dim3 mixGrid(33, 32);
  const dim3 invGrid(SLICES, 5);

  k_build_basis<<<1, 256, 0, stream>>>(Bas, BasInvT);
  k_encode<<<(NPIX + 255) / 256, 256, 0, stream>>>(x, 3, 3 * 65536, src_w, src_b, A);
  k_encode<<<(NPIX + 255) / 256, 256, 0, stream>>>(ug, 5, 0, geo_w, geo_b, G);

  auto spectral = [&](float* T, const float* b1, const float* b2) {
    k_dft_fwd<<<SLICES, 512, 0, stream>>>(T, Bas, S1, 0);
    k_mix<<<mixGrid, 512, 0, stream>>>(S1, MixY, S2);
    k_dft_inv<<<invGrid, 512, 0, stream>>>(S2, BasInvT, Z, 0, 0);
    k_dft_fwd<<<SLICES, 512, 0, stream>>>(T, Bas, S1, 1);
    k_mix<<<mixGrid, 512, 0, stream>>>(S1, MixX, S2);
    k_dft_inv<<<invGrid, 512, 0, stream>>>(S2, BasInvT, Z, 1, 1);
    k_mlp<<<NPIX / 64, 512, 0, stream>>>(Z, M1T, b1, M2T, b2, T);
  };

  for (int l = 0; l < 2; ++l) {
    k_pack_mix<<<32, 256, 0, stream>>>(front_wy + (long)l * 262144, MixY);
    k_pack_mix<<<32, 256, 0, stream>>>(front_wx + (long)l * 262144, MixX);
    k_pack_wT<<<32, 256, 0, stream>>>(front_m1w + (long)l * 8192, M1T, 64, 128);
    k_pack_wT<<<32, 256, 0, stream>>>(front_m2w + (long)l * 8192, M2T, 128, 64);
    spectral(A, front_m1b + l * 128, front_m2b + l * 64);
    spectral(G, front_m1b + l * 128, front_m2b + l * 64);
  }

  k_pack_wT<<<48, 256, 0, stream>>>(fuse_w, FWT, 192, 64);
  k_fuse<<<NPIX / 64, 512, 0, stream>>>(A, G, FWT, fuse_b, A);

  for (int l = 0; l < 2; ++l) {
    k_pack_mix<<<32, 256, 0, stream>>>(back_wy + (long)l * 262144, MixY);
    k_pack_mix<<<32, 256, 0, stream>>>(back_wx + (long)l * 262144, MixX);
    k_pack_wT<<<32, 256, 0, stream>>>(back_m1w + (long)l * 8192, M1T, 64, 128);
    k_pack_wT<<<32, 256, 0, stream>>>(back_m2w + (long)l * 8192, M2T, 128, 64);
    spectral(A, back_m1b + l * 128, back_m2b + l * 64);
  }

  k_project<<<(4 * 256 * 256 + 255) / 256, 256, 0, stream>>>(A, proj_w, proj_b, (float*)d_out);
  (void)in_sizes; (void)n_in; (void)out_size; (void)ws_size;
}
